// DecoderLayer_11974368821579
// MI455X (gfx1250) — compile-verified, matrix operand order checked by reading, not run on hardware
//
#include <hip/hip_runtime.h>
#include <hip/hip_bf16.h>

// ---------------- CDNA5 WMMA types ----------------
typedef __attribute__((ext_vector_type(16))) __bf16    v16bf;
typedef __attribute__((ext_vector_type(8)))  float     v8f;
typedef __attribute__((ext_vector_type(8)))  unsigned  v8u;

__device__ __forceinline__ unsigned short f2bf(float f) {
    unsigned u = __builtin_bit_cast(unsigned, f);
    unsigned r = u + 0x7FFFu + ((u >> 16) & 1u);   // round-to-nearest-even
    return (unsigned short)(r >> 16);
}

__device__ __forceinline__ v8f v8f_zero() {
    v8f z;
#pragma unroll
    for (int i = 0; i < 8; ++i) z[i] = 0.0f;
    return z;
}

__device__ __forceinline__ v8f wmma_bf16(v8u a, v8u b, v8f c) {
    return __builtin_amdgcn_wmma_f32_16x16x32_bf16(
        false, __builtin_bit_cast(v16bf, a),
        false, __builtin_bit_cast(v16bf, b),
        (short)0, c, false, false);
}

// K offset inside a 16x32 bf16 A/B fragment for vgpr-dword j, lane-half hl.
// (ISA 7.12.2: VGPR0..3 -> K 0..7 (+8*half), VGPR4..7 -> K 16..23 (+8*half))
__device__ __forceinline__ int frag_kk(int j, int hl) {
    return ((j & 3) << 1) + ((j >> 2) << 4) + (hl << 3);
}

// ---------------- constants ----------------
#define BB   2
#define NN   2048
#define DD   256
#define HH   4
#define DHH  64
#define NSPLIT 4
#define KCHUNK (NN / NSPLIT)   // 512 keys per split wave

// ---------------- weight pre-pack: fp32 row-major -> bf16 K-pair dwords ----
// out[(k/2)*Ncols + n] = { bf16 W[k][n] , bf16 W[k+1][n] }
__global__ void pack_w(const float* __restrict__ W, unsigned* __restrict__ out,
                       int Ncols, int total) {
    int i = blockIdx.x * blockDim.x + threadIdx.x;
    if (i >= total) return;
    int kp = i / Ncols, n = i - kp * Ncols;
    unsigned lo = f2bf(W[(2 * kp) * Ncols + n]);
    unsigned hi = f2bf(W[(2 * kp + 1) * Ncols + n]);
    out[i] = lo | (hi << 16);
}

// ---------------- LayerNorm (one block per row of 256) ----------------
__global__ void ln_kernel(const float* __restrict__ x, const float* __restrict__ g,
                          const float* __restrict__ be, float* __restrict__ outf,
                          unsigned short* __restrict__ outb) {
    __shared__ float red[256];
    int row = blockIdx.x, t = threadIdx.x;
    float v = x[row * DD + t];
    red[t] = v; __syncthreads();
#pragma unroll
    for (int s = 128; s > 0; s >>= 1) { if (t < s) red[t] += red[t + s]; __syncthreads(); }
    float mu = red[0] * (1.0f / DD); __syncthreads();
    float dv = v - mu;
    red[t] = dv * dv; __syncthreads();
#pragma unroll
    for (int s = 128; s > 0; s >>= 1) { if (t < s) red[t] += red[t + s]; __syncthreads(); }
    float var = red[0] * (1.0f / DD);
    float o = dv * __frsqrt_rn(var + 1e-5f) * g[t] + be[t];
    if (outf) outf[row * DD + t] = o;
    outb[row * DD + t] = f2bf(o);
}

// ------- generic wave-tile WMMA GEMM, NT column-tiles per wave -------------
// mode 0: out bf16 = acc + bias                    (Q, K projections)
// mode 1: out bf16 transposed [b,h,d,token]        (V projection)
// mode 2: out f32  = acc + bias + res              (x1 = xn + attnproj)
// mode 3: out bf16 = gelu(acc + bias)              (FFN up)
// mode 4: out f32  = acc + bias + res              (final output)
template <int NT>
__global__ __launch_bounds__(256)
void gemm_wmma(const unsigned short* __restrict__ A, const unsigned* __restrict__ Bp,
               const float* __restrict__ bias, const float* __restrict__ res,
               void* __restrict__ outp, int M, int Ncols, int K, int mode) {
    int lane = threadIdx.x & 31;
    int wid  = (blockIdx.x * blockDim.x + threadIdx.x) >> 5;
    int tiles_n = Ncols / (16 * NT);
    if (wid >= (M >> 4) * tiles_n) return;
    int tm = wid / tiles_n, tn = wid - tm * tiles_n;
    int row0 = tm << 4, col0 = tn * 16 * NT;
    int hl = lane >> 4, ln16 = lane & 15;

    const unsigned* A32 = (const unsigned*)A;
    v8f acc[NT];
#pragma unroll
    for (int c = 0; c < NT; ++c) acc[c] = v8f_zero();
    int arow = row0 + ln16;

    for (int k0 = 0; k0 < K; k0 += 32) {
        v8u av;
#pragma unroll
        for (int j = 0; j < 8; ++j)
            av[j] = A32[(arow * K + k0 + frag_kk(j, hl)) >> 1];
        // prefetch next weight K-slab (emits global_prefetch_b8)
        __builtin_prefetch(&Bp[((k0 + 32) >> 1) * Ncols + col0 + ln16], 0, 1);
#pragma unroll
        for (int c = 0; c < NT; ++c) {
            v8u bv;
#pragma unroll
            for (int j = 0; j < 8; ++j)
                bv[j] = Bp[((k0 + frag_kk(j, hl)) >> 1) * Ncols + col0 + c * 16 + ln16];
            acc[c] = wmma_bf16(av, bv, acc[c]);
        }
    }

    unsigned short* ob = (unsigned short*)outp;
    float*          of = (float*)outp;
#pragma unroll
    for (int c = 0; c < NT; ++c)
#pragma unroll
        for (int e = 0; e < 8; ++e) {
            int row = row0 + e + 8 * hl;
            int col = col0 + c * 16 + ln16;
            float v = acc[c][e] + bias[col];
            if (mode == 0) {
                ob[row * Ncols + col] = f2bf(v);
            } else if (mode == 1) {   // transposed V: [(b*H+h)*DH + d] * N + token
                int b  = row >> 11, nt = row & (NN - 1);
                int h  = col >> 6,  d  = col & (DHH - 1);
                ob[(((b * HH + h) * DHH) + d) * NN + nt] = f2bf(v);
            } else if (mode == 2) {
                of[row * Ncols + col] = v + res[row * Ncols + col];
            } else if (mode == 3) {
                float gl = 0.5f * v * (1.0f + erff(v * 0.70710678118654752f));
                ob[row * Ncols + col] = f2bf(gl);
            } else {
                of[row * Ncols + col] = v + res[row * Ncols + col];
            }
        }
}

// ------- flash attention, split-K: 4 waves per (b,h,16-query tile) ---------
// Block = 256 threads = 8 waves = 2 q-tiles x 4 key-splits. Each split wave
// runs online softmax over its 512 keys, partials merged in LDS.
__global__ __launch_bounds__(256)
void attn_kernel(const unsigned short* __restrict__ qb, const unsigned short* __restrict__ kb,
                 const unsigned short* __restrict__ vT, const float* __restrict__ coords,
                 const unsigned char* __restrict__ pmask, const float* __restrict__ semb,
                 const float* __restrict__ temb, unsigned short* __restrict__ a_out) {
    __shared__ float    accS[2][NSPLIT][16][DHH];   // 32 KB partial outputs
    __shared__ float    mS[2][NSPLIT][16];
    __shared__ float    lS[2][NSPLIT][16];
    __shared__ unsigned pbuf[8 * 256];              // 1KB bf16 P-tile per wave
    int lane = threadIdx.x & 31;
    int wib  = threadIdx.x >> 5;
    int ws   = wib & (NSPLIT - 1);                  // key split id
    int ql   = wib >> 2;                            // local q-tile (0/1)
    int gid  = blockIdx.x * 2 + ql;                 // global (b,h,qtile) id
    int qt = gid & 127;
    int h  = (gid >> 7) & (HH - 1);
    int b  = gid >> 9;
    int hl = lane >> 4, ln16 = lane & 15;

    const unsigned* q32 = (const unsigned*)qb;
    const unsigned* k32 = (const unsigned*)kb;
    const unsigned* v32 = (const unsigned*)vT;
    unsigned short* ps  = (unsigned short*)&pbuf[wib * 256];

    // Q fragments for head-dim 0..63 (two 16x32 bf16 A tiles)
    v8u qa[2];
    int qrowA = b * NN + qt * 16 + ln16;
#pragma unroll
    for (int f = 0; f < 2; ++f)
#pragma unroll
        for (int j = 0; j < 8; ++j)
            qa[f][j] = q32[(qrowA * DD + h * DHH + f * 32 + frag_kk(j, hl)) >> 1];

    // per-lane query coords for the 8 C-layout rows (e + 8*half)
    float tq[8], sqx[8], sqy[8], mrow[8], lrow[8];
    v8f acc[4];
#pragma unroll
    for (int e = 0; e < 8; ++e) {
        int r = b * NN + qt * 16 + e + 8 * hl;
        tq[e] = coords[r * 3]; sqx[e] = coords[r * 3 + 1]; sqy[e] = coords[r * 3 + 2];
        mrow[e] = -3.0e38f; lrow[e] = 0.0f;
    }
#pragma unroll
    for (int c = 0; c < 4; ++c) acc[c] = v8f_zero();

    for (int kb0 = ws * KCHUNK; kb0 < (ws + 1) * KCHUNK; kb0 += 32) {
        v8f sc[2];
#pragma unroll
        for (int c = 0; c < 2; ++c) {
            sc[c] = v8f_zero();
            int key = kb0 + c * 16 + ln16;
#pragma unroll
            for (int f = 0; f < 2; ++f) {
                v8u kf;
#pragma unroll
                for (int j = 0; j < 8; ++j)
                    kf[j] = k32[((b * NN + key) * DD + h * DHH + f * 32 + frag_kk(j, hl)) >> 1];
                sc[c] = wmma_bf16(qa[f], kf, sc[c]);
            }
            float tk  = coords[(b * NN + key) * 3];
            float skx = coords[(b * NN + key) * 3 + 1];
            float sky = coords[(b * NN + key) * 3 + 2];
            int   pm  = pmask[b * NN + key];
#pragma unroll
            for (int e = 0; e < 8; ++e) {
                float dt = tq[e] - tk;
                float dx = sqx[e] - skx, dy = sqy[e] - sky;
                float sd = __fsqrt_rn(dx * dx + dy * dy);
                int sidx = (int)(sd * 0.125f); if (sidx > 31) sidx = 31;
                int tidx = (int)fminf(fmaxf(dt + 16.0f, 0.0f), 32.0f);
                float bias = semb[sidx * HH + h] + temb[tidx * HH + h];
                float val = sc[c][e] * 0.125f + bias;   // 1/sqrt(64)
                if (sd > 256.0f || pm) val = -1.0e30f;
                sc[c][e] = val;
            }
        }
        // online softmax across the 32 new columns (16 lanes per half-group)
#pragma unroll
        for (int e = 0; e < 8; ++e) {
            float rm = fmaxf(sc[0][e], sc[1][e]);
            rm = fmaxf(rm, __shfl_xor(rm, 1)); rm = fmaxf(rm, __shfl_xor(rm, 2));
            rm = fmaxf(rm, __shfl_xor(rm, 4)); rm = fmaxf(rm, __shfl_xor(rm, 8));
            float newm  = fmaxf(mrow[e], rm);
            float scale = __expf(mrow[e] - newm);
            float p0 = __expf(sc[0][e] - newm);
            float p1 = __expf(sc[1][e] - newm);
            float rs = p0 + p1;
            rs += __shfl_xor(rs, 1); rs += __shfl_xor(rs, 2);
            rs += __shfl_xor(rs, 4); rs += __shfl_xor(rs, 8);
            lrow[e] = lrow[e] * scale + rs;
            mrow[e] = newm;
#pragma unroll
            for (int c = 0; c < 4; ++c) acc[c][e] *= scale;
            ps[(e + 8 * hl) * 32 + ln16]      = f2bf(p0);
            ps[(e + 8 * hl) * 32 + 16 + ln16] = f2bf(p1);
        }
        asm volatile("s_wait_dscnt 0" ::: "memory");   // cross-lane LDS RAW
        __builtin_amdgcn_wave_barrier();
        v8u pf;
#pragma unroll
        for (int j = 0; j < 8; ++j)
            pf[j] = pbuf[wib * 256 + ln16 * 16 + (frag_kk(j, hl) >> 1)];
#pragma unroll
        for (int c = 0; c < 4; ++c) {
            v8u vf;
#pragma unroll
            for (int j = 0; j < 8; ++j)
                vf[j] = v32[(((b * HH + h) * DHH + c * 16 + ln16) * NN + kb0 + frag_kk(j, hl)) >> 1];
            acc[c] = wmma_bf16(pf, vf, acc[c]);
        }
        __builtin_amdgcn_wave_barrier();               // keep P loads before next stores
    }

    // write per-split partial state to LDS
#pragma unroll
    for (int e = 0; e < 8; ++e) {
        int row = e + 8 * hl;
        if (ln16 == 0) { mS[ql][ws][row] = mrow[e]; lS[ql][ws][row] = lrow[e]; }
#pragma unroll
        for (int c = 0; c < 4; ++c)
            accS[ql][ws][row][c * 16 + ln16] = acc[c][e];
    }
    __syncthreads();

    // cooperative merge: 256 threads x 8 output elements each
    int t   = threadIdx.x;
    int ql2 = t >> 7;                       // 1024 outputs per q-tile / 8 = 128 threads
    int rem = (t & 127) * 8;                // element offset inside q-tile
    int row = rem >> 6;                     // 0..15
    int c0  = rem & 63;                     // 0..56 step 8
    int gid2 = blockIdx.x * 2 + ql2;
    int qt2 = gid2 & 127, h2 = (gid2 >> 7) & (HH - 1), b2 = gid2 >> 9;
    float M = mS[ql2][0][row];
#pragma unroll
    for (int s = 1; s < NSPLIT; ++s) M = fmaxf(M, mS[ql2][s][row]);
    float w[NSPLIT], L = 0.0f;
#pragma unroll
    for (int s = 0; s < NSPLIT; ++s) {
        w[s] = __expf(mS[ql2][s][row] - M);
        L += w[s] * lS[ql2][s][row];
    }
    float inv = 1.0f / L;
#pragma unroll
    for (int cc = 0; cc < 8; ++cc) {
        float v = 0.0f;
#pragma unroll
        for (int s = 0; s < NSPLIT; ++s) v += w[s] * accS[ql2][s][row][c0 + cc];
        a_out[(b2 * NN + qt2 * 16 + row) * DD + h2 * DHH + c0 + cc] = f2bf(v * inv);
    }
}

// ---------------- host launcher ----------------
extern "C" void kernel_launch(void* const* d_in, const int* in_sizes, int n_in,
                              void* d_out, int out_size, void* d_ws, size_t ws_size,
                              hipStream_t stream) {
    const float* x      = (const float*)d_in[0];
    const float* y      = (const float*)d_in[1];
    const float* coords = (const float*)d_in[2];
    const unsigned char* pmask = (const unsigned char*)d_in[3];
    const float* Wq = (const float*)d_in[4];  const float* bq = (const float*)d_in[5];
    const float* Wk = (const float*)d_in[6];  const float* bk = (const float*)d_in[7];
    const float* Wv = (const float*)d_in[8];  const float* bv = (const float*)d_in[9];
    const float* Wc = (const float*)d_in[10]; const float* bc = (const float*)d_in[11];
    const float* W1 = (const float*)d_in[12]; const float* b1 = (const float*)d_in[13];
    const float* W2 = (const float*)d_in[14]; const float* b2 = (const float*)d_in[15];
    const float* g1 = (const float*)d_in[16]; const float* be1 = (const float*)d_in[17];
    const float* g2 = (const float*)d_in[18]; const float* be2 = (const float*)d_in[19];
    const float* g3 = (const float*)d_in[20]; const float* be3 = (const float*)d_in[21];
    const float* semb = (const float*)d_in[22];
    const float* temb = (const float*)d_in[23];

    const int M = BB * NN;                               // 4096 rows
    size_t o = 0;
    auto nxt = [&](size_t bytes) -> void* {
        void* p = (char*)d_ws + o; o += (bytes + 255) & ~(size_t)255; return p;
    };
    float*          xn_f = (float*)         nxt((size_t)M * DD * 4);
    unsigned short* xn_b = (unsigned short*)nxt((size_t)M * DD * 2);
    unsigned short* yn_b = (unsigned short*)nxt((size_t)M * DD * 2);
    unsigned short* q_b  = (unsigned short*)nxt((size_t)M * DD * 2);
    unsigned short* k_b  = (unsigned short*)nxt((size_t)M * DD * 2);
    unsigned short* vT_b = (unsigned short*)nxt((size_t)M * DD * 2);
    unsigned short* a_b  = (unsigned short*)nxt((size_t)M * DD * 2);
    float*          x1_f = (float*)         nxt((size_t)M * DD * 4);
    unsigned short* l3_b = (unsigned short*)nxt((size_t)M * DD * 2);
    unsigned short* h_b  = (unsigned short*)nxt((size_t)M * 4 * DD * 2);
    unsigned* Wq_p = (unsigned*)nxt((size_t)(DD / 2) * DD * 4);
    unsigned* Wk_p = (unsigned*)nxt((size_t)(DD / 2) * DD * 4);
    unsigned* Wv_p = (unsigned*)nxt((size_t)(DD / 2) * DD * 4);
    unsigned* Wc_p = (unsigned*)nxt((size_t)(DD / 2) * DD * 4);
    unsigned* W1_p = (unsigned*)nxt((size_t)(DD / 2) * 4 * DD * 4);
    unsigned* W2_p = (unsigned*)nxt((size_t)(4 * DD / 2) * DD * 4);

    // weight packing
    int tdd  = (DD / 2) * DD;        // 32768
    int tdd4 = (DD / 2) * 4 * DD;    // 131072
    pack_w<<<(tdd + 255) / 256, 256, 0, stream>>>(Wq, Wq_p, DD, tdd);
    pack_w<<<(tdd + 255) / 256, 256, 0, stream>>>(Wk, Wk_p, DD, tdd);
    pack_w<<<(tdd + 255) / 256, 256, 0, stream>>>(Wv, Wv_p, DD, tdd);
    pack_w<<<(tdd + 255) / 256, 256, 0, stream>>>(Wc, Wc_p, DD, tdd);
    pack_w<<<(tdd4 + 255) / 256, 256, 0, stream>>>(W1, W1_p, 4 * DD, tdd4);
    pack_w<<<(tdd4 + 255) / 256, 256, 0, stream>>>(W2, W2_p, DD, tdd4);

    // LayerNorms of inputs
    ln_kernel<<<M, 256, 0, stream>>>(x, g1, be1, xn_f, xn_b);
    ln_kernel<<<M, 256, 0, stream>>>(y, g2, be2, nullptr, yn_b);

    auto gblk = [](int m, int nc, int nt) { return (m / 16) * (nc / (16 * nt)) / 8; };
    // QKV projections (NT=2 -> 2048 waves each)
    gemm_wmma<2><<<gblk(M, DD, 2), 256, 0, stream>>>(xn_b, Wq_p, bq, nullptr, q_b,  M, DD, DD, 0);
    gemm_wmma<2><<<gblk(M, DD, 2), 256, 0, stream>>>(yn_b, Wk_p, bk, nullptr, k_b,  M, DD, DD, 0);
    gemm_wmma<2><<<gblk(M, DD, 2), 256, 0, stream>>>(yn_b, Wv_p, bv, nullptr, vT_b, M, DD, DD, 1);

    // flash attention, split-K (4096 waves)
    attn_kernel<<<512, 256, 0, stream>>>(q_b, k_b, vT_b, coords, pmask, semb, temb, a_b);

    // output projection + residual, LN, FFN
    gemm_wmma<2><<<gblk(M, DD, 2), 256, 0, stream>>>(a_b, Wc_p, bc, xn_f, x1_f, M, DD, DD, 2);
    ln_kernel<<<M, 256, 0, stream>>>(x1_f, g3, be3, nullptr, l3_b);
    gemm_wmma<4><<<gblk(M, 4 * DD, 4), 256, 0, stream>>>(l3_b, W1_p, b1, nullptr, h_b, M, 4 * DD, DD, 3);
    gemm_wmma<2><<<gblk(M, DD, 2), 256, 0, stream>>>(h_b, W2_p, b2, x1_f, (float*)d_out, M, DD, 4 * DD, 4);
}
